// SAE_57105885168101
// MI455X (gfx1250) — matrix operand order repaired and checked by reading, not compiled, not run on hardware
//
#include <hip/hip_runtime.h>
#include <hip/hip_bf16.h>

typedef __attribute__((ext_vector_type(16))) __bf16 v16bf;
typedef __attribute__((ext_vector_type(8)))  __bf16 v8bf;
typedef __attribute__((ext_vector_type(8)))  float  v8f;

#define KTOP 32
#define NPART 4                      // top-K partitions per row
#define ASTRIDE 264                  // 256 + 8 bf16 pad (4 dwords) -> conflict-free b128 reads

// ---------------------------------------------------------------------------
// Kernel 0: convert sae_in = x - b_dec and W_enc to bf16 in workspace
// ---------------------------------------------------------------------------
__global__ void sae_prep_convert(const float* __restrict__ x,
                                 const float* __restrict__ b_dec,
                                 const float* __restrict__ W_enc,
                                 __bf16* __restrict__ xb,
                                 __bf16* __restrict__ wb,
                                 int BD, int HD, int D) {
    int i = blockIdx.x * blockDim.x + threadIdx.x;
    if (i < BD) xb[i] = (__bf16)(x[i] - b_dec[i % D]);
    if (i < HD) wb[i] = (__bf16)W_enc[i];
}

// ---------------------------------------------------------------------------
// Kernel 1: zero accumulators (counts, colsum, {sum x^2, sum e^2})
// ---------------------------------------------------------------------------
__global__ void sae_zero_ws(float* __restrict__ counts, float* __restrict__ colsum,
                            float* __restrict__ scal, int H, int D) {
    int i = blockIdx.x * blockDim.x + threadIdx.x;
    if (i < H) counts[i] = 0.0f;
    if (i < D) colsum[i] = 0.0f;
    if (i < 2) scal[i]   = 0.0f;
}

// ---------------------------------------------------------------------------
// Kernel 2: per-column sums of x and global sum(x^2) (for total_variance)
// ---------------------------------------------------------------------------
__global__ void sae_stats(const float* __restrict__ x, float* __restrict__ colsum,
                          float* __restrict__ scal, int D, int rowsPerBlk) {
    int d  = threadIdx.x;
    int r0 = blockIdx.x * rowsPerBlk;
    float cs = 0.0f, sq = 0.0f;
    for (int r = 0; r < rowsPerBlk; ++r) {
        float v = x[(size_t)(r0 + r) * D + d];
        cs += v; sq += v * v;
    }
    atomicAdd(&colsum[d], cs);
    __shared__ float red[1024];
    red[d] = sq;
    __syncthreads();
    for (int s = blockDim.x >> 1; s > 0; s >>= 1) {
        if (d < s) red[d] += red[d + s];
        __syncthreads();
    }
    if (d == 0) atomicAdd(&scal[0], red[0]);
}

// ---------------------------------------------------------------------------
// Kernel 3: fused bf16-WMMA encoder GEMM + bias + ReLU + streaming top-K.
// One workgroup (8 waves) owns 32 rows; A tile staged in LDS (no reg spills),
// fragments rebuilt per K-chunk with ds_load_b128; W streams from L2.
// ---------------------------------------------------------------------------
__launch_bounds__(256)
__global__ void sae_encoder_topk(const __bf16* __restrict__ xb,
                                 const __bf16* __restrict__ wb,
                                 const float*  __restrict__ b_enc,
                                 float* __restrict__ topv,
                                 int*   __restrict__ topi,
                                 float* __restrict__ counts,
                                 int D, int H) {
    const int t    = threadIdx.x;
    const int lane = t & 31;
    const int wave = t >> 5;
    const int m    = lane & 15;       // row-in-tile for A/B, col N for C/D
    const int half = lane >> 4;       // K half-select
    const int row0 = blockIdx.x * 32;

    __shared__ __bf16         atile[32 * ASTRIDE];        // ~16.5 KB
    __shared__ float          stage[32 * 128];            // 16 KB
    __shared__ float          pv[32][NPART][KTOP];        // 16 KB
    __shared__ unsigned short pi[32][NPART][KTOP];        // 8 KB

    // ---- stage the 32 x D bf16 A tile into LDS (padded rows) ----
    {
        const int nChunks = 32 * (D / 8);                 // v8bf chunks
        for (int c = t; c < nChunks; c += 256) {
            int r  = c / (D / 8);
            int c8 = (c % (D / 8)) * 8;
            *(v8bf*)&atile[r * ASTRIDE + c8] =
                *(const v8bf*)(xb + (size_t)(row0 + r) * D + c8);
        }
    }

    // ---- init private partial top-K lists: thread t<128 owns (row, part) ----
    float minV = -1.0f; int minP = 0;
    if (t < 128) {
        int r = t >> 2, p = t & 3;
        for (int i = 0; i < KTOP; ++i) { pv[r][p][i] = -1.0f; pi[r][p][i] = 0; }
    }
    __syncthreads();

    const int NKC = D / 32;                               // 8
    const __bf16* arow0 = &atile[m * ASTRIDE];
    const __bf16* arow1 = &atile[(16 + m) * ASTRIDE];

    for (int hBase = 0; hBase < H; hBase += 128) {
        const int h0 = hBase + wave * 16;
        v8f acc0 = {}, acc1 = {};
        const __bf16* wrow = wb + (size_t)(h0 + m) * D + half * 16; // B: 16 contiguous K
        #pragma unroll
        for (int kc = 0; kc < NKC; ++kc) {
            const int d0 = kc * 32 + half * 16;           // e0..7 -> K = half*16+0..7
            v8bf lo0 = *(const v8bf*)(arow0 + d0);
            v8bf hi0 = *(const v8bf*)(arow0 + d0 + 8);    // e8..15 -> K = half*16+8..15
            v16bf fA = __builtin_shufflevector(lo0, hi0,
                0,1,2,3,4,5,6,7,8,9,10,11,12,13,14,15);
            v8bf lo1 = *(const v8bf*)(arow1 + d0);
            v8bf hi1 = *(const v8bf*)(arow1 + d0 + 8);
            v16bf fB = __builtin_shufflevector(lo1, hi1,
                0,1,2,3,4,5,6,7,8,9,10,11,12,13,14,15);
            v16bf bfrag = *(const v16bf*)(wrow + kc * 32);
            acc0 = __builtin_amdgcn_wmma_f32_16x16x32_bf16(false, fA, false, bfrag,
                                                           (short)0, acc0, false, false);
            acc1 = __builtin_amdgcn_wmma_f32_16x16x32_bf16(false, fB, false, bfrag,
                                                           (short)0, acc1, false, false);
        }
        if (hBase + 128 < H)
            __builtin_prefetch(wb + (size_t)(h0 + 128 + m) * D, 0, 1);

        __syncthreads();                 // previous chunk's scanners done with stage
        // bias + ReLU + stage (C/D layout: lane L -> col N=L%16, VGPR r -> row r+8*half)
        const int col  = wave * 16 + m;
        const float bias = b_enc[hBase + col];
        #pragma unroll
        for (int r = 0; r < 8; ++r) {
            float v0 = acc0[r] + bias; v0 = v0 > 0.0f ? v0 : 0.0f;
            stage[(r + 8 * half) * 128 + col] = v0;
            float v1 = acc1[r] + bias; v1 = v1 > 0.0f ? v1 : 0.0f;
            stage[(16 + r + 8 * half) * 128 + col] = v1;
        }
        __syncthreads();

        // streaming top-K update: thread t<128 owns (row r, partition p) -> 32 cands
        if (t < 128) {
            int r = t >> 2, p = t & 3;
            const float* srow = &stage[r * 128 + p * 32];
            for (int j = 0; j < 32; ++j) {
                float v = srow[j];
                if (v > minV) {
                    pv[r][p][minP] = v;
                    pi[r][p][minP] = (unsigned short)(hBase + p * 32 + j);
                    float mn = pv[r][p][0]; int mp = 0;
                    #pragma unroll
                    for (int i = 1; i < KTOP; ++i) {
                        float q = pv[r][p][i];
                        if (q < mn) { mn = q; mp = i; }
                    }
                    minV = mn; minP = mp;
                }
            }
        }
    }
    __syncthreads();

    // ---- merge NPART partial lists per row into final top-K, emit ----
    if (t < 32) {
        const int r = t;
        float*          fv = &pv[r][0][0];
        unsigned short* fi = &pi[r][0][0];
        float mn = fv[0]; int mp = 0;
        for (int i = 1; i < KTOP; ++i) { if (fv[i] < mn) { mn = fv[i]; mp = i; } }
        for (int p = 1; p < NPART; ++p) {
            for (int i = 0; i < KTOP; ++i) {
                float v = pv[r][p][i];
                if (v > mn) {
                    fv[mp] = v; fi[mp] = pi[r][p][i];
                    mn = fv[0]; mp = 0;
                    for (int q = 1; q < KTOP; ++q) {
                        if (fv[q] < mn) { mn = fv[q]; mp = q; }
                    }
                }
            }
        }
        const size_t ob = (size_t)(row0 + r) * KTOP;
        for (int i = 0; i < KTOP; ++i) {
            int h = (int)fi[i];
            topv[ob + i] = fv[i];
            topi[ob + i] = h;
            atomicAdd(&counts[h], 1.0f);
        }
    }
}

// ---------------------------------------------------------------------------
// Kernel 4: sparse decode x_rec = z @ W_dec + b_dec, plus sum(e^2)
// ---------------------------------------------------------------------------
__global__ void sae_decode(const float* __restrict__ x,
                           const float* __restrict__ W_dec,
                           const float* __restrict__ b_dec,
                           const float* __restrict__ topv,
                           const int*   __restrict__ topi,
                           float* __restrict__ out,
                           float* __restrict__ scal, int D) {
    const int b = blockIdx.x;
    const int d = threadIdx.x;
    float acc = b_dec[d];
    const size_t tb = (size_t)b * KTOP;
    #pragma unroll 4
    for (int k = 0; k < KTOP; ++k) {
        float v = topv[tb + k];
        int   h = topi[tb + k];
        acc += v * W_dec[(size_t)h * D + d];
    }
    const size_t o = (size_t)b * D + d;
    out[o] = acc;
    float e = acc - x[o];
    __shared__ float red[1024];
    red[d] = e * e;
    __syncthreads();
    for (int s = blockDim.x >> 1; s > 0; s >>= 1) {
        if (d < s) red[d] += red[d + s];
        __syncthreads();
    }
    if (d == 0) atomicAdd(&scal[1], red[0]);
}

// ---------------------------------------------------------------------------
// Kernel 5: emit counts and fvu into d_out tail
// ---------------------------------------------------------------------------
__global__ void sae_finalize(const float* __restrict__ counts,
                             const float* __restrict__ colsum,
                             const float* __restrict__ scal,
                             float* __restrict__ out,
                             int B, int D, int H, size_t countsOff) {
    int i = blockIdx.x * blockDim.x + threadIdx.x;
    if (i < H) out[countsOff + i] = counts[i];
    if (i == 0) {
        float cs2 = 0.0f;
        for (int d = 0; d < D; ++d) cs2 += colsum[d] * colsum[d];
        float total_variance = scal[0] - cs2 / (float)B;
        out[countsOff + H] = scal[1] / total_variance;
    }
}

// ---------------------------------------------------------------------------
extern "C" void kernel_launch(void* const* d_in, const int* in_sizes, int n_in,
                              void* d_out, int out_size, void* d_ws, size_t ws_size,
                              hipStream_t stream) {
    const float* x     = (const float*)d_in[0];
    const float* W_enc = (const float*)d_in[1];
    const float* b_enc = (const float*)d_in[2];
    const float* W_dec = (const float*)d_in[3];
    const float* b_dec = (const float*)d_in[4];
    // d_in[5] holds k on device; K=32 is compiled in (KTOP).

    const int D = in_sizes[4];          // 256
    const int H = in_sizes[2];          // 8192
    const int B = in_sizes[0] / D;      // 16384
    float* out = (float*)d_out;

    // -------- workspace carve-out (256B-aligned regions) --------
    char* ws = (char*)d_ws;
    size_t off = 0;
    auto alloc = [&](size_t bytes) -> void* {
        void* p = ws + off;
        off += (bytes + 255) & ~(size_t)255;
        return p;
    };
    __bf16* xb     = (__bf16*)alloc((size_t)B * D * sizeof(__bf16));
    __bf16* wb     = (__bf16*)alloc((size_t)H * D * sizeof(__bf16));
    float*  topv   = (float*) alloc((size_t)B * KTOP * sizeof(float));
    int*    topi   = (int*)   alloc((size_t)B * KTOP * sizeof(int));
    float*  counts = (float*) alloc((size_t)H * sizeof(float));
    float*  colsum = (float*) alloc((size_t)D * sizeof(float));
    float*  scal   = (float*) alloc(2 * sizeof(float));
    (void)ws_size; (void)n_in; (void)out_size;

    const int BD = B * D, HD = H * D;
    const int mx = BD > HD ? BD : HD;

    sae_prep_convert<<<(mx + 255) / 256, 256, 0, stream>>>(x, b_dec, W_enc, xb, wb, BD, HD, D);

    const int zn = (H > D ? H : D);
    sae_zero_ws<<<(zn + 255) / 256, 256, 0, stream>>>(counts, colsum, scal, H, D);

    const int rowsPerBlk = 256;
    sae_stats<<<B / rowsPerBlk, D, 0, stream>>>(x, colsum, scal, D, rowsPerBlk);

    sae_encoder_topk<<<B / 32, 256, 0, stream>>>(xb, wb, b_enc, topv, topi, counts, D, H);

    sae_decode<<<B, D, 0, stream>>>(x, W_dec, b_dec, topv, topi, out, scal, D);

    sae_finalize<<<(H + 255) / 256, 256, 0, stream>>>(counts, colsum, scal, out,
                                                      B, D, H, (size_t)B * D);
}
